// CNNRNN_86973087744329
// MI455X (gfx1250) — compile-verified, matrix operand order checked by reading, not run on hardware
//
#include <hip/hip_runtime.h>
#include <hip/hip_bf16.h>

// ---------------------------------------------------------------------------
// Problem sizes (fixed by the reference)
// ---------------------------------------------------------------------------
static constexpr int BB = 128;    // batch
static constexpr int LL = 1024;   // sequence length
static constexpr int VV = 512;    // vocab
static constexpr int HH = 512;    // hidden
static constexpr int CC = 20;     // classes
static constexpr int G3 = 3 * HH; // 1536 (r,z,n gates)

typedef __attribute__((ext_vector_type(16))) _Float16 v16h;
typedef __attribute__((ext_vector_type(8)))  float    v8f;
typedef __attribute__((ext_vector_type(4)))  int      v4i;

struct __align__(16) h8 { _Float16 v[8]; };

// ---------------------------------------------------------------------------
// gfx1250 async global->LDS path (ASYNCcnt-tracked). Fall back to plain
// generic stores into shared (ds_store_b128) if the builtin is absent.
// Builtin signature (from hipcc diagnostics): (v4i __device__*, v4i __shared__*,
// imm offset, imm cpol).
// ---------------------------------------------------------------------------
#if defined(__has_builtin)
#  if __has_builtin(__builtin_amdgcn_global_load_async_to_lds_b128)
#    define HAS_ASYNC_LDS 1
#  endif
#endif
#ifndef HAS_ASYNC_LDS
#  define HAS_ASYNC_LDS 0
#endif

__device__ __forceinline__ void copy16_g2l(_Float16* lds_dst, const _Float16* gsrc) {
#if HAS_ASYNC_LDS
  __builtin_amdgcn_global_load_async_to_lds_b128(
      (__attribute__((address_space(1))) v4i*)gsrc,
      (__attribute__((address_space(3))) v4i*)lds_dst,
      0, 0);
#else
  *(h8*)lds_dst = *(const h8*)gsrc;
#endif
}

__device__ __forceinline__ void wait_async_all() {
#if HAS_ASYNC_LDS
  asm volatile("s_wait_asynccnt 0x0" ::: "memory");
#endif
}
__device__ __forceinline__ void wait_async_le2() {
#if HAS_ASYNC_LDS
  asm volatile("s_wait_asynccnt 0x2" ::: "memory");
#endif
}

// ---------------------------------------------------------------------------
// WMMA helpers (CDNA5: v_wmma_f32_16x16x32_f16, wave32)
// 16-bit A/B fragment for a row-major source with stride ld:
//   lane L: row = row0 + (L & 15)
//   halves[0..7]  = k0 + 8*(L>>4) .. +7        (one 16B load)
//   halves[8..15] = k0 + 16 + 8*(L>>4) .. +7   (one 16B load)
// Works for global or LDS base pointers (LDS provenance -> ds_load_b128).
// ---------------------------------------------------------------------------
__device__ __forceinline__ v16h load_frag(const _Float16* base,
                                          int row0, int k0, int ld) {
  const int lane = threadIdx.x & 31;
  const _Float16* p = base + (size_t)(row0 + (lane & 15)) * ld + k0 + ((lane >> 4) << 3);
  union { v16h v; h8 q[2]; } u;
  u.q[0] = *(const h8*)(p);
  u.q[1] = *(const h8*)(p + 16);
  return u.v;
}

__device__ __forceinline__ v8f wmma_f16(v16h a, v16h b, v8f c) {
  // (neg_a, A, neg_b, B, c_mod, C, reuse_a, reuse_b)
  return __builtin_amdgcn_wmma_f32_16x16x32_f16(false, a, false, b, (short)0, c,
                                                false, false);
}

// ---------------------------------------------------------------------------
// Kernel 1: prep — f32->f16 weight conversion, conv-weight transpose to
// [k][v][h] (coalesced gather), h-state zero init, barrier init.
// ---------------------------------------------------------------------------
__global__ void prep_kernel(const float* __restrict__ cw,     // (H,V,3)
                            const float* __restrict__ wih,    // (3H,H)
                            const float* __restrict__ whh,    // (3H,H)
                            _Float16* __restrict__ cwT,       // (3,V,H) f16
                            _Float16* __restrict__ wih16,
                            _Float16* __restrict__ whh16,
                            float* __restrict__ h32,
                            _Float16* __restrict__ h16,
                            unsigned* __restrict__ bar) {
  const int idx = blockIdx.x * 256 + threadIdx.x;   // grid covers 786432
  if (idx < 3 * VV * HH) {
    const int h = idx & (HH - 1);
    const int t = idx >> 9;            // k*V + v
    const int v = t & (VV - 1);
    const int k = t >> 9;              // 0..2
    cwT[idx]   = (_Float16)cw[((size_t)h * VV + v) * 3 + k];
    wih16[idx] = (_Float16)wih[idx];
    whh16[idx] = (_Float16)whh[idx];
  }
  if (idx < BB * HH) { h32[idx] = 0.f; h16[idx] = (_Float16)0.f; }
  if (idx < 2) bar[idx] = 0u;
}

// ---------------------------------------------------------------------------
// Kernel 2: embed — Conv1d over one-hot == 3 gathers. Time-major output
// y[(l*B+b), h] in f16 so step-t rows are contiguous for the GRU.
// ---------------------------------------------------------------------------
__global__ void embed_kernel(const int* __restrict__ x,       // (B,L)
                             const _Float16* __restrict__ cwT,
                             const float* __restrict__ cb,
                             _Float16* __restrict__ y) {
  const int row = blockIdx.x;          // l*B + b
  const int l = row >> 7;
  const int b = row & (BB - 1);
  const int i0 = (l > 0)      ? x[b * LL + l - 1] : -1;
  const int i1 =                x[b * LL + l];
  const int i2 = (l < LL - 1) ? x[b * LL + l + 1] : -1;
  const _Float16* w0 = cwT;
  const _Float16* w1 = cwT + (size_t)VV * HH;
  const _Float16* w2 = cwT + (size_t)2 * VV * HH;
  for (int h = threadIdx.x; h < HH; h += 256) {
    float acc = cb[h];
    if (i0 >= 0) acc += (float)w0[(size_t)i0 * HH + h];
    acc += (float)w1[(size_t)i1 * HH + h];
    if (i2 >= 0) acc += (float)w2[(size_t)i2 * HH + h];
    y[(size_t)row * HH + h] = (_Float16)fmaxf(acc, 0.f);
  }
}

// ---------------------------------------------------------------------------
// Kernel 3: gi GEMM — gi = relu_out @ W_ih^T + b_ih, all timesteps at once.
// M=131072, K=512, N=1536. Block tile 128x128 (8 waves, each 64x32).
// B k-tiles staged to LDS with async global->LDS copies, double buffered:
// tile ks+2 is issued right after the barrier so its ASYNCcnt copy overlaps
// the WMMA of tile ks / ks+1.  A streams straight from global (L2-resident).
// ---------------------------------------------------------------------------
static constexpr int BLD = 40;  // 32 halves + 8 pad -> 80B row stride (16B ok)

__global__ void __launch_bounds__(256)
gi_gemm_kernel(const _Float16* __restrict__ y,
               const _Float16* __restrict__ wih16,
               const float* __restrict__ bih,
               _Float16* __restrict__ gi) {
  __shared__ _Float16 bsh[2][128 * BLD];   // 2 x 10 KB

  const int wave = threadIdx.x >> 5, lane = threadIdx.x & 31;
  const int m0w  = blockIdx.x * 128 + (wave & 1) * 64;   // 4 m-tiles of 16
  const int nloc = (wave >> 1) * 32;                     // 2 n-tiles of 16
  const int n0b  = blockIdx.y * 128;
  const _Float16* bsrc = wih16 + (size_t)n0b * HH;

  // stage k-tile ks (32 halves per each of the 128 n-rows) into bsh[ks&1]
  auto stage = [&](int ks) {
    const int k0 = ks * 32;
    _Float16* dst = bsh[ks & 1];
    for (int c = threadIdx.x; c < 512; c += 256) {       // 2 chunks / thread
      const int r = c >> 2, c8 = (c & 3) << 3;
      copy16_g2l(dst + r * BLD + c8, bsrc + (size_t)r * HH + k0 + c8);
    }
  };

  v8f acc[4][2];
  const v8f vz = {0.f, 0.f, 0.f, 0.f, 0.f, 0.f, 0.f, 0.f};
#pragma unroll
  for (int i = 0; i < 4; ++i)
#pragma unroll
    for (int j = 0; j < 2; ++j) acc[i][j] = vz;

  stage(0);
  stage(1);
  for (int ks = 0; ks < HH / 32; ++ks) {
    wait_async_le2();        // my 2 copies for tile ks are done (in-order)
    __syncthreads();         // everyone's copies for tile ks are done
    const int k0 = ks * 32;
    const _Float16* bt = bsh[ks & 1];
    v16h a[4], bb[2];
#pragma unroll
    for (int i = 0; i < 4; ++i) a[i] = load_frag(y, m0w + 16 * i, k0, HH);
#pragma unroll
    for (int j = 0; j < 2; ++j) bb[j] = load_frag(bt, nloc + 16 * j, 0, BLD);
#pragma unroll
    for (int i = 0; i < 4; ++i)
#pragma unroll
      for (int j = 0; j < 2; ++j) acc[i][j] = wmma_f16(a[i], bb[j], acc[i][j]);
    __syncthreads();         // all reads of bsh[ks&1] done before reuse
    if (ks + 2 < HH / 32) stage(ks + 2);
  }

  const int hi = lane >> 4, nl = lane & 15;
#pragma unroll
  for (int j = 0; j < 2; ++j) {
    const int n = n0b + nloc + 16 * j + nl;
    const float bias = bih[n];
#pragma unroll
    for (int i = 0; i < 4; ++i)
#pragma unroll
      for (int q = 0; q < 8; ++q) {
        const int row = m0w + 16 * i + q + 8 * hi;
        gi[(size_t)row * G3 + n] = (_Float16)(acc[i][j][q] + bias);
      }
  }
}

// ---------------------------------------------------------------------------
// Device-wide split barrier for the persistent GRU kernel.
// s_cluster_barrier is a NOP when not dispatched as a cluster (ISA §3.2) and
// a real HW barrier when clustered; the atomic arrive/release makes it
// correct for a plain 24-WG dispatch. State reset by prep_kernel each launch.
// ---------------------------------------------------------------------------
__device__ __forceinline__ void grid_sync(unsigned* bar, unsigned nb,
                                          unsigned& my_gen) {
  __builtin_amdgcn_s_cluster_barrier();
  __syncthreads();
  if (threadIdx.x == 0) {
    const unsigned arrived =
        __hip_atomic_fetch_add(&bar[0], 1u, __ATOMIC_ACQ_REL, __HIP_MEMORY_SCOPE_AGENT);
    if (arrived == nb - 1u) {
      __hip_atomic_store(&bar[0], 0u, __ATOMIC_RELAXED, __HIP_MEMORY_SCOPE_AGENT);
      __hip_atomic_fetch_add(&bar[1], 1u, __ATOMIC_RELEASE, __HIP_MEMORY_SCOPE_AGENT);
    } else {
      while (__hip_atomic_load(&bar[1], __ATOMIC_ACQUIRE, __HIP_MEMORY_SCOPE_AGENT) == my_gen)
        __builtin_amdgcn_s_sleep(8);
    }
  }
  ++my_gen;
  __syncthreads();
}

// ---------------------------------------------------------------------------
// Kernel 4: GRU recurrence, persistent (24 WGs).
// The WG's time-invariant 64x512 W_hh slice (64 KB) is staged into LDS ONCE
// via async global->LDS before the time loop -> per-step B fragments come
// from ds_load_b128 at LDS latency instead of L2 latency (this sits on the
// sequential critical path).  h (128 KB) stays global/L2 (changes per step).
// Phase A: gh = h @ W_hh^T + b_hh via WMMA (each WG a 128x64 stripe).
// Phase B: gate math + h update in f32, store f16 copy for next step's A.
// ---------------------------------------------------------------------------
__global__ void __launch_bounds__(256)
gru_kernel(const _Float16* __restrict__ gi,
           const _Float16* __restrict__ whh16,
           const float* __restrict__ bhh,
           float* __restrict__ gh,
           float* __restrict__ h32,
           _Float16* __restrict__ h16,
           unsigned* __restrict__ bar) {
  __shared__ _Float16 wsh[64 * HH];              // 64 KB, stride HH

  const int tid = threadIdx.x;
  const unsigned nb = gridDim.x;                 // 24
  const int wave = tid >> 5, lane = tid & 31;
  const int m0w  = (wave >> 1) * 32;             // 4 m-stripes of 32
  const int nloc = (wave & 1) * 32;              // within the 64-row slice
  const int n0b  = blockIdx.x * 64;
  const int hi = lane >> 4, nl = lane & 15;
  const int gtid = blockIdx.x * 256 + tid;
  const int nthreads = (int)nb * 256;
  unsigned my_gen = 0;

  // one-time W_hh slice staging (contiguous 64 KB)
  {
    const _Float16* bsrc = whh16 + (size_t)n0b * HH;
    for (int c = tid; c < (64 * HH) / 8; c += 256)   // 16 chunks / thread
      copy16_g2l(&wsh[c * 8], bsrc + (size_t)c * 8);
    wait_async_all();
    __syncthreads();
  }

  const v8f vz = {0.f, 0.f, 0.f, 0.f, 0.f, 0.f, 0.f, 0.f};

  for (int t = 0; t < LL; ++t) {
    // ---- Phase A: gh stripe via WMMA (B from LDS) -------------------------
    v8f acc[2][2];
#pragma unroll
    for (int i = 0; i < 2; ++i)
#pragma unroll
      for (int j = 0; j < 2; ++j) acc[i][j] = vz;

#pragma unroll 4
    for (int k0 = 0; k0 < HH; k0 += 32) {
      v16h a0 = load_frag(h16, m0w, k0, HH);
      v16h a1 = load_frag(h16, m0w + 16, k0, HH);
      v16h b0 = load_frag(wsh, nloc, k0, HH);
      v16h b1 = load_frag(wsh, nloc + 16, k0, HH);
      acc[0][0] = wmma_f16(a0, b0, acc[0][0]);
      acc[0][1] = wmma_f16(a0, b1, acc[0][1]);
      acc[1][0] = wmma_f16(a1, b0, acc[1][0]);
      acc[1][1] = wmma_f16(a1, b1, acc[1][1]);
    }

    // warm L2 with this step's gi rows while the barrier settles
    __builtin_prefetch(gi + (size_t)t * BB * G3 + (size_t)gtid * 32, 0, 0);

#pragma unroll
    for (int j = 0; j < 2; ++j) {
      const int n = n0b + nloc + 16 * j + nl;
      const float bias = bhh[n];
#pragma unroll
      for (int i = 0; i < 2; ++i)
#pragma unroll
        for (int q = 0; q < 8; ++q) {
          const int row = m0w + 16 * i + q + 8 * hi;
          gh[(size_t)row * G3 + n] = acc[i][j][q] + bias;
        }
    }
    __threadfence();
    grid_sync(bar, nb, my_gen);

    // ---- Phase B: gates + state update ------------------------------------
    const size_t gi_t = (size_t)t * BB * G3;
    for (int idx = gtid; idx < BB * HH; idx += nthreads) {
      const int b = idx >> 9;
      const int j = idx & (HH - 1);
      const size_t gb = gi_t + (size_t)b * G3 + j;
      const float ir = (float)gi[gb];
      const float iz = (float)gi[gb + HH];
      const float in = (float)gi[gb + 2 * HH];
      const size_t hb = (size_t)b * G3 + j;
      const float hr = gh[hb];
      const float hz = gh[hb + HH];
      const float hn = gh[hb + 2 * HH];
      const float r = 1.f / (1.f + __expf(-(ir + hr)));
      const float z = 1.f / (1.f + __expf(-(iz + hz)));
      const float n = tanhf(in + r * hn);
      const float hnew = (1.f - z) * n + z * h32[idx];
      h32[idx] = hnew;
      h16[idx] = (_Float16)hnew;
    }
    __threadfence();
    grid_sync(bar, nb, my_gen);
  }
}

// ---------------------------------------------------------------------------
// Kernel 5: classifier — tiny (128x20x512), plain f32 dot products.
// ---------------------------------------------------------------------------
__global__ void cls_kernel(const float* __restrict__ h,
                           const float* __restrict__ w,
                           const float* __restrict__ bias,
                           float* __restrict__ out) {
  const int idx = blockIdx.x * 256 + threadIdx.x;
  if (idx >= BB * CC) return;
  const int b = idx / CC, c = idx % CC;
  const float* hp = h + (size_t)b * HH;
  const float* wp = w + (size_t)c * HH;
  float acc = bias[c];
  for (int j = 0; j < HH; ++j) acc = fmaf(hp[j], wp[j], acc);
  out[idx] = acc;
}

// ---------------------------------------------------------------------------
// Host launcher
// ---------------------------------------------------------------------------
extern "C" void kernel_launch(void* const* d_in, const int* in_sizes, int n_in,
                              void* d_out, int out_size, void* d_ws, size_t ws_size,
                              hipStream_t stream) {
  (void)in_sizes; (void)n_in; (void)out_size; (void)ws_size;
  const int*   x      = (const int*)  d_in[0];
  const float* conv_w = (const float*)d_in[1];
  const float* conv_b = (const float*)d_in[2];
  const float* w_ih   = (const float*)d_in[3];
  const float* w_hh   = (const float*)d_in[4];
  const float* b_ih   = (const float*)d_in[5];
  const float* b_hh   = (const float*)d_in[6];
  const float* cls_w  = (const float*)d_in[7];
  const float* cls_b  = (const float*)d_in[8];
  float* out = (float*)d_out;

  // workspace carve-up (256B aligned)
  char* ws = (char*)d_ws;
  size_t off = 0;
  auto carve = [&](size_t bytes) {
    char* p = ws + off;
    off = (off + bytes + 255) & ~(size_t)255;
    return p;
  };
  _Float16* cwT   = (_Float16*)carve((size_t)3 * VV * HH * 2);  // 1.5 MB
  _Float16* wih16 = (_Float16*)carve((size_t)G3 * HH * 2);      // 1.5 MB
  _Float16* whh16 = (_Float16*)carve((size_t)G3 * HH * 2);      // 1.5 MB
  _Float16* y16   = (_Float16*)carve((size_t)LL * BB * HH * 2); // 128 MB
  _Float16* gi16  = (_Float16*)carve((size_t)LL * BB * G3 * 2); // 402 MB
  float*    gh32  = (float*)   carve((size_t)BB * G3 * 4);      // 768 KB
  float*    h32   = (float*)   carve((size_t)BB * HH * 4);      // 256 KB
  _Float16* h16   = (_Float16*)carve((size_t)BB * HH * 2);      // 128 KB
  unsigned* bar   = (unsigned*)carve(256);

  // 1) weight prep + state/barrier init
  prep_kernel<<<(3 * VV * HH + 255) / 256, 256, 0, stream>>>(
      conv_w, w_ih, w_hh, cwT, wih16, whh16, h32, h16, bar);

  // 2) conv-as-gather + ReLU, time-major f16
  embed_kernel<<<LL * BB, 256, 0, stream>>>(x, cwT, conv_b, y16);

  // 3) all-timesteps input-gate GEMM (WMMA, async-LDS staged weights)
  gi_gemm_kernel<<<dim3((LL * BB) / 128, G3 / 128, 1), 256, 0, stream>>>(
      y16, wih16, b_ih, gi16);

  // 4) persistent GRU recurrence (WMMA, LDS-resident weights, grid barrier)
  gru_kernel<<<G3 / 64, 256, 0, stream>>>(gi16, whh16, b_hh, gh32, h32, h16, bar);

  // 5) classifier
  cls_kernel<<<(BB * CC + 255) / 256, 256, 0, stream>>>(h32, cls_w, cls_b, out);
}
